// groupSSM_23708219473991
// MI455X (gfx1250) — matrix operand order repaired and checked by reading, not compiled, not run on hardware
//
#include <hip/hip_runtime.h>
#include <cstdint>
#include <cstddef>

#define DEVFN __device__ __forceinline__

typedef _Float16 half16 __attribute__((ext_vector_type(16)));
typedef float    float8 __attribute__((ext_vector_type(8)));

constexpr int G_ = 4, B_ = 8, H_ = 64, W_ = 64, L_ = 4096;
constexpr int DM_ = 128, D_ = 32, DI_ = 64, N_ = 16, K_ = 4, R_ = 2, C_ = R_ + 2 * N_; // 34

// workspace layout (float elements)
constexpr size_t OFF_XZ = 0;                               // (G,B,128,L) in_proj output: e<64 = conv input, e>=64 = z
constexpr size_t SZ_XZ  = (size_t)G_ * B_ * 128 * L_;
constexpr size_t OFF_XC = OFF_XZ + SZ_XZ;                  // (G,B,64,L) conv+SiLU output
constexpr size_t SZ_XC  = (size_t)G_ * B_ * DI_ * L_;
constexpr size_t OFF_XD = OFF_XC + SZ_XC;                  // (G,B,K,34,L) x_dbl (dts|Bs|Cs)
constexpr size_t SZ_XD  = (size_t)G_ * B_ * K_ * C_ * L_;
constexpr size_t OFF_Y  = OFF_XD + SZ_XD;                  // (G,B,64,L) combined scan output (row-major hw order)

// ---- WMMA helpers (CDNA5 v_wmma_f32_16x16x32_f16) ----------------------------
DEVFN float8 wmma_f16(half16 a, half16 b, float8 c) {
  return __builtin_amdgcn_wmma_f32_16x16x32_f16(false, a, false, b, (short)0, c, false, false);
}
// A k order used here (contiguous, not the ISA table): lane lo holds row M=lo; K index per packed half j:
// K = j + 16*hi   (hi = lane>>4), matching the B fills
DEVFN int a_kidx(int j, int hi) { return j + 16 * hi; }

// direction map: scan index l (for direction k) -> row-major spatial position
DEVFN int dirmap(int k, int l) {
  int t = (k & 2) ? (L_ - 1 - l) : l;
  return (k & 1) ? ((t & (H_ - 1)) * W_ + (t >> 6)) : t;
}

// ---- gfx1250 async global->LDS copy ------------------------------------------
// Pass the real LDS element pointer: its low 32 bits are the wave-relative LDS
// byte address (ISA 10.2 aperture rules), and the ptrtoint escape makes the
// "memory" clobber actually alias the __shared__ array for the compiler.
DEVFN void async_copy_f32(float* lds_ptr, const float* gptr) {
  asm volatile("global_load_async_to_lds_b32 %0, %1, off"
               :: "v"((unsigned)(uintptr_t)lds_ptr),
                  "v"((unsigned long long)(uintptr_t)gptr)
               : "memory");
}
DEVFN void wait_async0() { asm volatile("s_wait_asynccnt 0" ::: "memory"); }

// ---- kernel 1: in_proj GEMM (per pixel: xz[e] = sum_d x[d]*W[e,d]) -----------
__global__ void __launch_bounds__(256) k_inproj(const float* __restrict__ x,
                                                const float* __restrict__ inw,
                                                float* __restrict__ ws) {
  float* xz = ws + OFF_XZ;
  const int wid = blockIdx.x * (blockDim.x >> 5) + (threadIdx.x >> 5);
  const int lane = threadIdx.x & 31, lo = lane & 15, hi = lane >> 4;
  const int tile = wid & 255;
  const int b = (wid >> 8) & 7;
  const int g = wid >> 11;
  const int pos = tile * 16 + lo;

  // B fragment: 32(d) x 16(pos) from x[b, g*32+d, h, w]
  half16 bf{};
#pragma unroll
  for (int j = 0; j < 16; ++j) {
    int d = j + 16 * hi;
    bf[j] = (_Float16)x[((size_t)(b * DM_ + g * D_ + d)) * L_ + pos];
  }
#pragma unroll
  for (int et = 0; et < 8; ++et) {
    half16 af{};
#pragma unroll
    for (int j = 0; j < 16; ++j)
      af[j] = (_Float16)inw[(g * 128 + et * 16 + lo) * 32 + a_kidx(j, hi)];
    float8 c{};
    c = wmma_f16(af, bf, c);
#pragma unroll
    for (int i = 0; i < 8; ++i) {
      int e = et * 16 + i + 8 * hi;
      xz[(((size_t)(g * B_ + b)) * 128 + e) * L_ + pos] = c[i];
    }
  }
}

// ---- kernel 2: depthwise 3x3 conv + bias + SiLU ------------------------------
__global__ void __launch_bounds__(256) k_conv(const float* __restrict__ cw,
                                              const float* __restrict__ cb,
                                              float* __restrict__ ws) {
  const float* xz = ws + OFF_XZ;
  float* xc = ws + OFF_XC;
  int idx = blockIdx.x * blockDim.x + threadIdx.x;  // G*B*DI*L
  int pos = idx & (L_ - 1);
  int d = (idx >> 12) & (DI_ - 1);
  int b = (idx >> 18) & (B_ - 1);
  int g = idx >> 21;
  int h = pos >> 6, w = pos & 63;
  const float* src = xz + ((size_t)(g * B_ + b) * 128 + d) * L_;
  const float* wk = cw + (g * DI_ + d) * 9;
  float acc = cb[g * DI_ + d];
#pragma unroll
  for (int kh = 0; kh < 3; ++kh) {
    int hh = h + kh - 1;
    if ((unsigned)hh >= 64u) continue;
#pragma unroll
    for (int kw = 0; kw < 3; ++kw) {
      int wp = w + kw - 1;
      if ((unsigned)wp >= 64u) continue;
      acc += src[hh * 64 + wp] * wk[kh * 3 + kw];
    }
  }
  xc[((size_t)(g * B_ + b) * DI_ + d) * L_ + pos] = acc / (1.0f + __expf(-acc));
}

// ---- kernel 3: x_dbl GEMM (34x64 weight times directional 64xL activations) --
__global__ void __launch_bounds__(256) k_xdbl(const float* __restrict__ xpw,
                                              float* __restrict__ ws) {
  const float* xc = ws + OFF_XC;
  float* xd = ws + OFF_XD;
  const int wid = blockIdx.x * (blockDim.x >> 5) + (threadIdx.x >> 5);
  const int lane = threadIdx.x & 31, lo = lane & 15, hi = lane >> 4;
  const int tile = wid & 255;
  const int kd = (wid >> 8) & 3;
  const int b = (wid >> 10) & 7;
  const int g = wid >> 13;
  const int pos0 = tile * 16;

  const float* xcb = xc + (size_t)(g * B_ + b) * DI_ * L_;
  half16 bf0{}, bf1{};  // two K-steps of 32 over d
#pragma unroll
  for (int j = 0; j < 16; ++j) {
    int dd = j + 16 * hi;
    int gp = dirmap(kd, pos0 + lo);
    bf0[j] = (_Float16)xcb[(size_t)dd * L_ + gp];
    bf1[j] = (_Float16)xcb[(size_t)(dd + 32) * L_ + gp];
  }
  float* outb = xd + (size_t)((g * B_ + b) * K_ + kd) * C_ * L_;
  const float* wb = xpw + (size_t)(g * K_ + kd) * C_ * DI_;
#pragma unroll
  for (int mt = 0; mt < 3; ++mt) {  // 3 row tiles cover 34 rows (zero-padded)
    half16 af0{}, af1{};
    int m = mt * 16 + lo;
#pragma unroll
    for (int j = 0; j < 16; ++j) {
      int kk = a_kidx(j, hi);
      af0[j] = (_Float16)((m < C_) ? wb[m * DI_ + kk] : 0.f);
      af1[j] = (_Float16)((m < C_) ? wb[m * DI_ + 32 + kk] : 0.f);
    }
    float8 c{};
    c = wmma_f16(af0, bf0, c);
    c = wmma_f16(af1, bf1, c);
#pragma unroll
    for (int i = 0; i < 8; ++i) {
      int row = mt * 16 + i + 8 * hi;
      if (row < C_) outb[(size_t)row * L_ + pos0 + lo] = c[i];
    }
  }
}

// ---- kernel 4: zero the combined y buffer (atomicAdd target) -----------------
__global__ void __launch_bounds__(256) k_zero(float* __restrict__ ws) {
  size_t i = (size_t)blockIdx.x * blockDim.x + threadIdx.x;
  (ws + OFF_Y)[i] = 0.f;
}

// ---- kernel 5: selective scan. 1 block per (g,b,k), thread = (d,n) state. ----
__global__ void __launch_bounds__(1024) k_scan(const float* __restrict__ dtw,
                                               const float* __restrict__ dtb,
                                               const float* __restrict__ alog,
                                               const float* __restrict__ Ds,
                                               float* __restrict__ ws) {
  const float* xc = ws + OFF_XC;
  const float* xd = ws + OFF_XD;
  float* y = ws + OFF_Y;
  const int kd = blockIdx.x & 3;
  const int b = (blockIdx.x >> 2) & 7;
  const int g = blockIdx.x >> 5;
  const int tid = threadIdx.x;
  const int d = tid >> 4, n = tid & 15;

  const float Av = -__expf(alog[(g * 256 + kd * 64 + d) * 16 + n]);
  const float dw0 = dtw[((g * 4 + kd) * 64 + d) * 2 + 0];
  const float dw1 = dtw[((g * 4 + kd) * 64 + d) * 2 + 1];
  const float db = dtb[(g * 4 + kd) * 64 + d];
  const float Dv = Ds[g * 256 + kd * 64 + d];

  const float* xdb = xd + (size_t)((g * B_ + b) * K_ + kd) * C_ * L_;
  const float* xcb = xc + (size_t)(g * B_ + b) * DI_ * L_;
  float* yb = y + (size_t)(g * B_ + b) * DI_ * L_;

  __shared__ float lds[(C_ + DI_) * 32];  // rows: 0-1 dts, 2-17 Bs, 18-33 Cs, 34-97 u
  float hst = 0.f;

  for (int chunk = 0; chunk < L_ / 32; ++chunk) {
    const int l0 = chunk * 32;
    // stage 32 steps of (dts,Bs,Cs,u) into LDS via gfx1250 async copies
    for (int i = tid; i < (C_ + DI_) * 32; i += 1024) {
      int row = i >> 5, c = i & 31, l = l0 + c;
      const float* src = (row < C_) ? (xdb + (size_t)row * L_ + l)
                                    : (xcb + (size_t)(row - C_) * L_ + dirmap(kd, l));
      async_copy_f32(&lds[i], src);
    }
    wait_async0();
    __syncthreads();
#pragma unroll 4
    for (int c = 0; c < 32; ++c) {
      float pre = lds[c] * dw0 + lds[32 + c] * dw1 + db;
      float dt = (pre > 20.f) ? pre : __logf(1.f + __expf(pre));  // softplus
      float bt = lds[(2 + n) * 32 + c];
      float ct = lds[(2 + N_ + n) * 32 + c];
      float u = lds[(C_ + d) * 32 + c];
      hst = hst * __expf(dt * Av) + (dt * bt) * u;
      float yp = hst * ct;
      yp += __shfl_xor(yp, 8, 16);
      yp += __shfl_xor(yp, 4, 16);
      yp += __shfl_xor(yp, 2, 16);
      yp += __shfl_xor(yp, 1, 16);
      if (n == 0)
        atomicAdd(&yb[(size_t)d * L_ + dirmap(kd, l0 + c)], yp + u * Dv);
    }
    __syncthreads();
  }
}

// ---- kernel 6: LayerNorm * SiLU(z), then out-proj GEMM (32x64) ---------------
__global__ void __launch_bounds__(256) k_out(const float* __restrict__ lng,
                                             const float* __restrict__ lnb,
                                             const float* __restrict__ ow,
                                             const float* __restrict__ ws,
                                             float* __restrict__ out) {
  const float* xz = ws + OFF_XZ;
  const float* y = ws + OFF_Y;
  const int wid = blockIdx.x * (blockDim.x >> 5) + (threadIdx.x >> 5);
  const int lane = threadIdx.x & 31, lo = lane & 15, hi = lane >> 4;
  const int tile = wid & 255;
  const int b = (wid >> 8) & 7;
  const int g = wid >> 11;
  const int pos = tile * 16 + lo;

  const float* yb = y + (size_t)(g * B_ + b) * DI_ * L_;
  const float* zb = xz + ((size_t)(g * B_ + b) * 128 + 64) * L_;

  // per-position stats over DI=64: each half-lane does 32, combine across halves
  float sum = 0.f, ss = 0.f;
#pragma unroll
  for (int e0 = 0; e0 < 32; ++e0) {
    float v = yb[(size_t)(hi * 32 + e0) * L_ + pos];
    sum += v;
    ss += v * v;
  }
  sum += __shfl_xor(sum, 16, 32);
  ss += __shfl_xor(ss, 16, 32);
  float mu = sum * (1.f / 64.f);
  float var = ss * (1.f / 64.f) - mu * mu;
  float rstd = rsqrtf(var + 1e-5f);

  half16 bf[2];
#pragma unroll
  for (int s = 0; s < 2; ++s) {
#pragma unroll
    for (int j = 0; j < 16; ++j) {
      int e = s * 32 + j + 16 * hi;
      float yv = yb[(size_t)e * L_ + pos];
      float zv = zb[(size_t)e * L_ + pos];
      float ym = ((yv - mu) * rstd * lng[g * 64 + e] + lnb[g * 64 + e]) *
                 (zv / (1.f + __expf(-zv)));
      bf[s][j] = (_Float16)ym;
    }
  }
#pragma unroll
  for (int mt = 0; mt < 2; ++mt) {
    half16 af0{}, af1{};
#pragma unroll
    for (int j = 0; j < 16; ++j) {
      int kk = a_kidx(j, hi);
      af0[j] = (_Float16)ow[(g * 32 + mt * 16 + lo) * 64 + kk];
      af1[j] = (_Float16)ow[(g * 32 + mt * 16 + lo) * 64 + 32 + kk];
    }
    float8 c{};
    c = wmma_f16(af0, bf[0], c);
    c = wmma_f16(af1, bf[1], c);
#pragma unroll
    for (int i = 0; i < 8; ++i) {
      int dd = mt * 16 + i + 8 * hi;
      out[((size_t)b * DM_ + g * D_ + dd) * L_ + tile * 16 + lo] = c[i];
    }
  }
}

extern "C" void kernel_launch(void* const* d_in, const int* in_sizes, int n_in,
                              void* d_out, int out_size, void* d_ws, size_t ws_size,
                              hipStream_t stream) {
  (void)in_sizes; (void)n_in; (void)out_size; (void)ws_size;
  const float* x = (const float*)d_in[0];
  const float* inw = (const float*)d_in[1];
  const float* cw = (const float*)d_in[2];
  const float* cb = (const float*)d_in[3];
  const float* xpw = (const float*)d_in[4];
  const float* dtw = (const float*)d_in[5];
  const float* dtb = (const float*)d_in[6];
  const float* alog = (const float*)d_in[7];
  const float* Ds = (const float*)d_in[8];
  const float* lng = (const float*)d_in[9];
  const float* lnb = (const float*)d_in[10];
  const float* ow = (const float*)d_in[11];
  float* ws = (float*)d_ws;
  float* out = (float*)d_out;

  k_inproj<<<1024, 256, 0, stream>>>(x, inw, ws);       // 8192 waves
  k_conv<<<32768, 256, 0, stream>>>(cw, cb, ws);        // 8.4M elems
  k_xdbl<<<4096, 256, 0, stream>>>(xpw, ws);            // 32768 waves
  k_zero<<<32768, 256, 0, stream>>>(ws);                // zero y
  k_scan<<<128, 1024, 0, stream>>>(dtw, dtb, alog, Ds, ws);
  k_out<<<1024, 256, 0, stream>>>(lng, lnb, ow, ws, out);
}
